// PTBModel_49864570306723
// MI455X (gfx1250) — matrix operand order repaired and checked by reading, not compiled, not run on hardware
//
#include <hip/hip_runtime.h>
#include <hip/hip_bf16.h>

// ---------------------------------------------------------------------------
// HyperLSTM for MI455X (gfx1250, wave32, WMMA).
// Strategy: bf16-convert + transpose weights once per launch (stay L2-resident,
// 192MB L2 >> 14MB weights), then per timestep run small M=64 WMMA GEMMs with
// v_wmma_f32_16x16x32_bf16, fragments loaded straight from global (L2).
// ---------------------------------------------------------------------------

typedef __bf16 bf16_t;
typedef __attribute__((ext_vector_type(16))) __bf16 v16bf;
typedef __attribute__((ext_vector_type(8)))  __bf16 v8bf;
typedef __attribute__((ext_vector_type(8)))  float   v8f;

#define V_  256
#define D_  256
#define H_  1024
#define HH_ 256
#define E_  64
#define T_  256
#define B_  64
#define EPS_ 1e-5f

__device__ __forceinline__ float sigf(float x) { return 1.0f / (1.0f + __expf(-x)); }

// ---- WMMA fragment loaders (C/D per CDNA5 ISA 7.12.2; A/B use contiguous k order) ------------------
// A: 16x32 bf16, row m = lane&15; lanes<16 hold K {0..15}, lanes>=16
//    hold K {16..31}. Two 16-byte contiguous loads per lane.
__device__ __forceinline__ v16bf load_a_frag(const bf16_t* __restrict__ A, int lda,
                                             int m0, int k0) {
  const int l  = threadIdx.x & 31;
  const int m  = m0 + (l & 15);
  const int kh = (l >> 4) << 4;
  const bf16_t* p = A + m * lda + k0 + kh;
  v8bf lo = *(const v8bf*)(p);        // K = kh .. kh+7
  v8bf hi = *(const v8bf*)(p + 8);    // K = kh+8 .. kh+15
  v16bf r;
#pragma unroll
  for (int i = 0; i < 8; ++i) { r[i] = lo[i]; r[i + 8] = hi[i]; }
  return r;
}

// B: 32x16 bf16 (KxN). Weights are stored transposed (N x K), so lane n=l&15
// reads 16 contiguous K values (one aligned 32-byte load):
// lanes<16 -> K k0..k0+15 ; lanes>=16 -> K k0+16..k0+31.
__device__ __forceinline__ v16bf load_b_frag(const bf16_t* __restrict__ Bt, int ldb,
                                             int n0, int k0) {
  const int l  = threadIdx.x & 31;
  const int n  = n0 + (l & 15);
  const int kk = k0 + ((l >> 4) << 4);
  return *(const v16bf*)(Bt + n * ldb + kk);
}

// C/D: VGPR r holds (M = m0 + r + 8*(lane>=16), N = n0 + lane&15).
__device__ __forceinline__ void store_c_tile(float* __restrict__ C, int ldc,
                                             int m0, int n0, v8f acc,
                                             const float* __restrict__ bias,
                                             bf16_t* __restrict__ Cbf, int ldcb) {
  const int l  = threadIdx.x & 31;
  const int n  = n0 + (l & 15);
  const int mb = m0 + ((l >> 4) << 3);
  const float bv = bias ? bias[n] : 0.0f;
#pragma unroll
  for (int r = 0; r < 8; ++r) {
    float v = acc[r] + bv;
    C[(mb + r) * ldc + n] = v;
    if (Cbf) Cbf[(mb + r) * ldcb + n] = (bf16_t)v;
  }
}

// ---- Generic GEMM: C(M x N) = A(M x K) @ Bt^T, Bt stored (N x K) bf16 ------
// grid.x = N/64 tiles, grid.y = M/64 tiles. 256 threads = 8 waves;
// wave w owns m-tile (w&3) and n-tiles {2*(w>>2), 2*(w>>2)+1}.
__global__ void gemm64_bf16_kernel(const bf16_t* __restrict__ A, int lda,
                                   const bf16_t* __restrict__ Bt, int K,
                                   float* __restrict__ C, int ldc,
                                   const float* __restrict__ bias,
                                   bf16_t* __restrict__ Cbf, int ldcb) {
  const int w   = threadIdx.x >> 5;
  const int mt  = w & 3;
  const int ntp = (w >> 2) << 1;
  const int m0  = blockIdx.y * 64 + mt * 16;
  const int n0  = blockIdx.x * 64 + ntp * 16;
  v8f acc0 = {}; v8f acc1 = {};
  for (int k = 0; k < K; k += 32) {
    v16bf a  = load_a_frag(A, lda, m0, k);
    v16bf b0 = load_b_frag(Bt, K, n0, k);
    v16bf b1 = load_b_frag(Bt, K, n0 + 16, k);
    acc0 = __builtin_amdgcn_wmma_f32_16x16x32_bf16(false, a, false, b0, (short)0, acc0, false, false);
    acc1 = __builtin_amdgcn_wmma_f32_16x16x32_bf16(false, a, false, b1, (short)0, acc1, false, false);
  }
  store_c_tile(C, ldc, m0, n0,      acc0, bias, Cbf, ldcb);
  store_c_tile(C, ldc, m0, n0 + 16, acc1, bias, Cbf, ldcb);
}

// ---- d_{h,x,b}[g] = z[:,kind,g,:] @ P_kind[g] : 12 GEMMs 64x64x1024 --------
// grid = (16, 4 gates, 3 kinds).
__global__ void gemm_d_kernel(const bf16_t* __restrict__ zbf,   // 64 x 768
                              const bf16_t* __restrict__ Pbf,   // 12 x (1024 x 64)
                              float* __restrict__ dbuf) {       // 3 x (64 x 4096)
  const int g    = blockIdx.y;
  const int kind = blockIdx.z;
  const bf16_t* A  = zbf + kind * 256 + g * 64;                 // lda=768, K=64
  const bf16_t* Bt = Pbf + (kind * 4 + g) * (1024 * 64);        // ldb=64
  float* C = dbuf + kind * (B_ * 4096) + g * 1024;              // ldc=4096
  const int w   = threadIdx.x >> 5;
  const int mt  = w & 3;
  const int ntp = (w >> 2) << 1;
  const int m0  = mt * 16;
  const int n0  = blockIdx.x * 64 + ntp * 16;
  v8f acc0 = {}; v8f acc1 = {};
  for (int k = 0; k < 64; k += 32) {
    v16bf a  = load_a_frag(A, 768, m0, k);
    v16bf b0 = load_b_frag(Bt, 64, n0, k);
    v16bf b1 = load_b_frag(Bt, 64, n0 + 16, k);
    acc0 = __builtin_amdgcn_wmma_f32_16x16x32_bf16(false, a, false, b0, (short)0, acc0, false, false);
    acc1 = __builtin_amdgcn_wmma_f32_16x16x32_bf16(false, a, false, b1, (short)0, acc1, false, false);
  }
  store_c_tile(C, 4096, m0, n0,      acc0, nullptr, nullptr, 0);
  store_c_tile(C, 4096, m0, n0 + 16, acc1, nullptr, nullptr, 0);
}

// ---- pointwise / LN kernels ------------------------------------------------
__device__ __forceinline__ void block_reduce2(float& s1, float& s2, float* sm) {
  const int tid = threadIdx.x;
  sm[tid] = s1; sm[256 + tid] = s2;
  __syncthreads();
  for (int st = 128; st > 0; st >>= 1) {
    if (tid < st) { sm[tid] += sm[tid + st]; sm[256 + tid] += sm[256 + tid + st]; }
    __syncthreads();
  }
  s1 = sm[0]; s2 = sm[256];
}

__global__ void hyper_cell_kernel(const float* __restrict__ hg,
                                  float* __restrict__ hc, float* __restrict__ hh,
                                  bf16_t* __restrict__ hh_bf) {
  const int b = blockIdx.x, j = threadIdx.x;
  const float* r = hg + b * 1024;
  float hi = r[j], hgt = r[256 + j], hf = r[512 + j], ho = r[768 + j];
  float hcn = sigf(hf) * hc[b * 256 + j] + sigf(hi) * tanhf(hgt);
  float hhn = sigf(ho) * tanhf(hcn);
  hc[b * 256 + j] = hcn;
  hh[b * 256 + j] = hhn;
  hh_bf[b * 256 + j] = (bf16_t)hhn;
}

__global__ void combine_ln_kernel(const float* __restrict__ dx, const float* __restrict__ dh,
                                  const float* __restrict__ db, const float* __restrict__ gx,
                                  const float* __restrict__ gh, const float* __restrict__ b0,
                                  const float* __restrict__ gamma, const float* __restrict__ beta,
                                  float* __restrict__ pre) {
  __shared__ float sm[512];
  const int g = blockIdx.x, b = blockIdx.y, tid = threadIdx.x;
  float v[4]; float s = 0.f, ss = 0.f;
#pragma unroll
  for (int i = 0; i < 4; ++i) {
    int n = tid + (i << 8);
    int col = g * 1024 + n;
    int idx = b * 4096 + col;
    float p = dx[idx] * gx[idx] + dh[idx] * gh[idx] + db[idx] + b0[col];
    v[i] = p; s += p; ss += p * p;
  }
  block_reduce2(s, ss, sm);
  float mean = s * (1.f / 1024.f);
  float var  = ss * (1.f / 1024.f) - mean * mean;
  float rstd = rsqrtf(var + EPS_);
#pragma unroll
  for (int i = 0; i < 4; ++i) {
    int n = tid + (i << 8);
    int col = g * 1024 + n;
    pre[b * 4096 + col] = (v[i] - mean) * rstd * gamma[col] + beta[col];
  }
}

__global__ void cell_update_kernel(const float* __restrict__ pre, float* __restrict__ c,
                                   float* __restrict__ h, bf16_t* __restrict__ h_bf,
                                   bf16_t* __restrict__ outs_bf_t,
                                   const float* __restrict__ gamma4,
                                   const float* __restrict__ beta4) {
  __shared__ float sm[512];
  const int b = blockIdx.x, tid = threadIdx.x;
  const float* pb = pre + b * 4096;
  float cn[4], ov[4]; float s = 0.f, ss = 0.f;
#pragma unroll
  for (int i = 0; i < 4; ++i) {
    int n = tid + (i << 8);
    float iv = pb[n], gv = pb[1024 + n], fv = pb[2048 + n];
    ov[i] = pb[3072 + n];
    float x = sigf(fv) * c[b * 1024 + n] + sigf(iv) * tanhf(gv);
    cn[i] = x; c[b * 1024 + n] = x;
    s += x; ss += x * x;
  }
  block_reduce2(s, ss, sm);
  float mean = s * (1.f / 1024.f);
  float var  = ss * (1.f / 1024.f) - mean * mean;
  float rstd = rsqrtf(var + EPS_);
#pragma unroll
  for (int i = 0; i < 4; ++i) {
    int n = tid + (i << 8);
    float cell = (cn[i] - mean) * rstd * gamma4[n] + beta4[n];
    float hn = sigf(ov[i]) * tanhf(cell);
    h[b * 1024 + n] = hn;
    h_bf[b * 1024 + n] = (bf16_t)hn;
    outs_bf_t[b * 1024 + n] = (bf16_t)hn;
  }
}

__global__ void build_acat_kernel(const bf16_t* __restrict__ xemb_t,
                                  const bf16_t* __restrict__ h_bf,
                                  const bf16_t* __restrict__ hh_bf,
                                  bf16_t* __restrict__ acat) {
  const int b = blockIdx.x, tid = threadIdx.x;
#pragma unroll
  for (int i = 0; i < 6; ++i) {
    int n = tid + (i << 8);
    bf16_t v;
    if (n < 256)       v = xemb_t[b * 256 + n];
    else if (n < 1280) v = h_bf[b * 1024 + (n - 256)];
    else               v = hh_bf[b * 256 + (n - 1280)];
    acat[b * 1536 + n] = v;
  }
}

__global__ void embed_gather_kernel(const int* __restrict__ inputs,
                                    const float* __restrict__ emb,
                                    bf16_t* __restrict__ xemb) {
  const int tb = blockIdx.x;
  const int idx = inputs[tb];
  xemb[tb * 256 + threadIdx.x] = (bf16_t)emb[idx * 256 + threadIdx.x];
}

__global__ void transpose_bf16_kernel(bf16_t* __restrict__ dst, int ldDst, int colOff,
                                      const float* __restrict__ src, int K, int N) {
  const int i = blockIdx.x * 256 + threadIdx.x;
  if (i >= K * N) return;
  const int k = i / N, n = i % N;
  dst[n * ldDst + colOff + k] = (bf16_t)src[i];
}

__global__ void convert_bf16_kernel(bf16_t* __restrict__ dst,
                                    const float* __restrict__ src, int n) {
  const int i = blockIdx.x * 256 + threadIdx.x;
  if (i < n) dst[i] = (bf16_t)src[i];
}

__global__ void zbias_kernel(float* __restrict__ zb, const float* __restrict__ bhz,
                             const float* __restrict__ bxz) {
  const int i = blockIdx.x * 256 + threadIdx.x;
  if (i < 256)      zb[i] = bhz[i];
  else if (i < 512) zb[i] = bxz[i - 256];
  else if (i < 768) zb[i] = 0.f;
}

__global__ void init_state_kernel(const float* __restrict__ h0, const float* __restrict__ c0,
                                  const float* __restrict__ hh0, const float* __restrict__ hc0,
                                  float* __restrict__ h, float* __restrict__ c,
                                  float* __restrict__ hh, float* __restrict__ hc,
                                  bf16_t* __restrict__ h_bf, bf16_t* __restrict__ hh_bf) {
  const int i = blockIdx.x * 256 + threadIdx.x;
  if (i < B_ * H_)  { h[i] = h0[i]; c[i] = c0[i]; h_bf[i] = (bf16_t)h0[i]; }
  if (i < B_ * HH_) { hh[i] = hh0[i]; hc[i] = hc0[i]; hh_bf[i] = (bf16_t)hh0[i]; }
}

__global__ void write_states_kernel(const float* __restrict__ h, const float* __restrict__ c,
                                    const float* __restrict__ hh, const float* __restrict__ hc,
                                    float* __restrict__ out) {
  const int i = blockIdx.x * 256 + threadIdx.x;
  const int L = T_ * B_ * V_;
  if (i < B_ * H_)  { out[L + i] = h[i]; out[L + B_ * H_ + i] = c[i]; }
  if (i < B_ * HH_) { out[L + 2 * B_ * H_ + i] = hh[i];
                      out[L + 2 * B_ * H_ + B_ * HH_ + i] = hc[i]; }
}

// ---------------------------------------------------------------------------
extern "C" void kernel_launch(void* const* d_in, const int* in_sizes, int n_in,
                              void* d_out, int out_size, void* d_ws, size_t ws_size,
                              hipStream_t stream) {
  (void)in_sizes; (void)n_in; (void)out_size; (void)ws_size;
  const int*   inputs    = (const int*)  d_in[0];
  const float* h0        = (const float*)d_in[1];
  const float* c0        = (const float*)d_in[2];
  const float* hh0       = (const float*)d_in[3];
  const float* hc0       = (const float*)d_in[4];
  const float* embedding = (const float*)d_in[5];
  const float* W_x       = (const float*)d_in[6];
  const float* W_h       = (const float*)d_in[7];
  const float* hyper_Wx  = (const float*)d_in[8];
  const float* hyper_Wh  = (const float*)d_in[9];
  const float* hyper_b   = (const float*)d_in[10];
  const float* W_hz      = (const float*)d_in[11];
  const float* b_hz      = (const float*)d_in[12];
  const float* P_h       = (const float*)d_in[13];
  const float* W_xz      = (const float*)d_in[14];
  const float* b_xz      = (const float*)d_in[15];
  const float* P_x       = (const float*)d_in[16];
  const float* W_bz      = (const float*)d_in[17];
  const float* P_b       = (const float*)d_in[18];
  const float* b0        = (const float*)d_in[19];
  const float* ln_gamma  = (const float*)d_in[20];
  const float* ln_beta   = (const float*)d_in[21];
  const float* out_W     = (const float*)d_in[22];
  const float* out_b     = (const float*)d_in[23];
  float* out = (float*)d_out;

  // workspace carve-out (256-byte aligned)
  char* base = (char*)d_ws; size_t off = 0;
  auto alloc = [&](size_t bytes) -> void* {
    off = (off + 255) & ~(size_t)255;
    void* p = base + off; off += bytes; return p;
  };
  bf16_t* Wx_bf   = (bf16_t*)alloc((size_t)4096 * 256 * 2);   // (N=4H x K=D)
  bf16_t* Wh_bf   = (bf16_t*)alloc((size_t)4096 * 1024 * 2);  // (4H x H)
  bf16_t* hyW_bf  = (bf16_t*)alloc((size_t)1024 * 1536 * 2);  // (4HH x D+H+HH)
  bf16_t* Wz_bf   = (bf16_t*)alloc((size_t)768 * 256 * 2);    // ([hz|xz|bz] x HH)
  bf16_t* P_bf    = (bf16_t*)alloc((size_t)12 * 1024 * 64 * 2);
  bf16_t* outW_bf = (bf16_t*)alloc((size_t)256 * 1024 * 2);   // (V x H) = already NxK
  float*  zbias   = (float*) alloc(768 * 4);
  bf16_t* xemb_bf = (bf16_t*)alloc((size_t)T_ * B_ * D_ * 2);
  bf16_t* outs_bf = (bf16_t*)alloc((size_t)T_ * B_ * H_ * 2);
  float*  h       = (float*) alloc((size_t)B_ * H_ * 4);
  float*  c       = (float*) alloc((size_t)B_ * H_ * 4);
  float*  hh      = (float*) alloc((size_t)B_ * HH_ * 4);
  float*  hc      = (float*) alloc((size_t)B_ * HH_ * 4);
  bf16_t* h_bf    = (bf16_t*)alloc((size_t)B_ * H_ * 2);
  bf16_t* hh_bf   = (bf16_t*)alloc((size_t)B_ * HH_ * 2);
  bf16_t* acat    = (bf16_t*)alloc((size_t)B_ * 1536 * 2);
  float*  hg      = (float*) alloc((size_t)B_ * 1024 * 4);
  float*  gx      = (float*) alloc((size_t)B_ * 4096 * 4);
  float*  gh      = (float*) alloc((size_t)B_ * 4096 * 4);
  float*  z       = (float*) alloc((size_t)B_ * 768 * 4);
  bf16_t* zbf     = (bf16_t*)alloc((size_t)B_ * 768 * 2);
  float*  dbuf    = (float*) alloc((size_t)3 * B_ * 4096 * 4);
  float*  pre     = (float*) alloc((size_t)B_ * 4096 * 4);

  // ---- one-time (per launch) weight prep ----
  auto tpose = [&](bf16_t* dst, int ldDst, int colOff, const float* src, int K, int N) {
    transpose_bf16_kernel<<<(K * N + 255) / 256, 256, 0, stream>>>(dst, ldDst, colOff, src, K, N);
  };
  tpose(Wx_bf, 256, 0, W_x, 256, 4096);
  tpose(Wh_bf, 1024, 0, W_h, 1024, 4096);
  tpose(hyW_bf, 1536, 0,    hyper_Wx, 1280, 1024);
  tpose(hyW_bf, 1536, 1280, hyper_Wh, 256, 1024);
  tpose(Wz_bf + 0 * 256 * 256, 256, 0, W_hz, 256, 256);
  tpose(Wz_bf + 1 * 256 * 256, 256, 0, W_xz, 256, 256);
  tpose(Wz_bf + 2 * 256 * 256, 256, 0, W_bz, 256, 256);
  const float* Psrc[3] = { P_h, P_x, P_b };
  for (int kind = 0; kind < 3; ++kind)
    for (int g = 0; g < 4; ++g)
      tpose(P_bf + (kind * 4 + g) * 65536, 64, 0, Psrc[kind] + (size_t)g * 64 * 1024, 64, 1024);
  convert_bf16_kernel<<<1024, 256, 0, stream>>>(outW_bf, out_W, 256 * 1024);
  zbias_kernel<<<3, 256, 0, stream>>>(zbias, b_hz, b_xz);
  embed_gather_kernel<<<T_ * B_, 256, 0, stream>>>(inputs, embedding, xemb_bf);
  init_state_kernel<<<256, 256, 0, stream>>>(h0, c0, hh0, hc0, h, c, hh, hc, h_bf, hh_bf);

  // ---- recurrence ----
  for (int t = 0; t < T_; ++t) {
    const bf16_t* xt = xemb_bf + (size_t)t * B_ * D_;
    build_acat_kernel<<<B_, 256, 0, stream>>>(xt, h_bf, hh_bf, acat);
    // hg = [x,h,hh] @ hyW + hyper_b            (64 x 1536 x 1024)
    gemm64_bf16_kernel<<<dim3(16, 1), 256, 0, stream>>>(acat, 1536, hyW_bf, 1536,
                                                        hg, 1024, hyper_b, nullptr, 0);
    // gx = x @ W_x                             (64 x 256 x 4096)
    gemm64_bf16_kernel<<<dim3(64, 1), 256, 0, stream>>>(xt, 256, Wx_bf, 256,
                                                        gx, 4096, nullptr, nullptr, 0);
    // gh = h @ W_h                             (64 x 1024 x 4096)
    gemm64_bf16_kernel<<<dim3(64, 1), 256, 0, stream>>>(h_bf, 1024, Wh_bf, 1024,
                                                        gh, 4096, nullptr, nullptr, 0);
    hyper_cell_kernel<<<B_, 256, 0, stream>>>(hg, hc, hh, hh_bf);
    // z = hh_n @ [W_hz|W_xz|W_bz] + [b_hz|b_xz|0]   (64 x 256 x 768)
    gemm64_bf16_kernel<<<dim3(12, 1), 256, 0, stream>>>(hh_bf, 256, Wz_bf, 256,
                                                        z, 768, zbias, zbf, 768);
    gemm_d_kernel<<<dim3(16, 4, 3), 256, 0, stream>>>(zbf, P_bf, dbuf);
    combine_ln_kernel<<<dim3(4, 64), 256, 0, stream>>>(
        dbuf + 1 * B_ * 4096, /* d_x */ dbuf, /* d_h */ dbuf + 2 * B_ * 4096, /* d_b */
        gx, gh, b0, ln_gamma, ln_beta, pre);
    cell_update_kernel<<<B_, 256, 0, stream>>>(pre, c, h, h_bf,
                                               outs_bf + (size_t)t * B_ * H_,
                                               ln_gamma + 4 * H_, ln_beta + 4 * H_);
  }

  // logits = outs @ out_W^T + out_b            (16384 x 1024 x 256)
  gemm64_bf16_kernel<<<dim3(4, 256), 256, 0, stream>>>(outs_bf, 1024, outW_bf, 1024,
                                                       out, 256, out_b, nullptr, 0);
  write_states_kernel<<<640, 256, 0, stream>>>(h, c, hh, hc, out);
}
